// PureGNN_34351148433711
// MI455X (gfx1250) — matrix operand order repaired and checked by reading, not compiled, not run on hardware
//
#include <hip/hip_runtime.h>
#include <hip/hip_bf16.h>
#include <math.h>

// ---------------------------------------------------------------------------
// PureGNN forward for MI455X (gfx1250, wave32).
//   h = x[:,None]*lin_w + lin_b
//   3x: h = scatter_add( (h @ W)[src]*norm, dst ) + b   (with self loops)
//   pooled = segment_sum(h, batch); out = tanh(pooled@r1+b1)@r2 + b2
// ---------------------------------------------------------------------------

#define N_NODES 50000
#define N_EDGES 1600000
#define E_DIM   128
#define N_GRAPHS 32

typedef __attribute__((ext_vector_type(16))) __bf16 v16bf;
typedef __attribute__((ext_vector_type(8)))  float  v8f;

// ---------------- degree / norm precompute --------------------------------

__global__ void k_fill1(float* __restrict__ deg, int n) {
    int i = blockIdx.x * blockDim.x + threadIdx.x;
    if (i < n) deg[i] = 1.0f;               // self-loop contributes 1 to deg
}

__global__ void k_deg(float* __restrict__ deg, const int* __restrict__ dst, int ne) {
    int e = blockIdx.x * blockDim.x + threadIdx.x;
    if (e < ne) unsafeAtomicAdd(&deg[dst[e]], 1.0f);
}

__global__ void k_rsqrt(float* __restrict__ d, int n) {
    int i = blockIdx.x * blockDim.x + threadIdx.x;
    if (i < n) {
        float v = d[i];
        d[i] = (v > 0.0f) ? __frsqrt_rn(v) : 0.0f;
    }
}

__global__ void k_norm(float* __restrict__ norm, const int* __restrict__ src,
                       const int* __restrict__ dst, const float* __restrict__ dinv, int ne) {
    int e = blockIdx.x * blockDim.x + threadIdx.x;
    if (e < ne) norm[e] = dinv[src[e]] * dinv[dst[e]];
}

// ---------------- input embedding -----------------------------------------

__global__ void k_embed(float* __restrict__ h, const float* __restrict__ x,
                        const float* __restrict__ w, const float* __restrict__ b) {
    int i = blockIdx.x * blockDim.x + threadIdx.x;   // over N_NODES*E_DIM
    int n = i >> 7, j = i & 127;
    h[i] = x[n] * w[j] + b[j];
}

// ---------------- W (f32 row-major) -> bf16 col-major ---------------------

__global__ void k_wcvt(__bf16* __restrict__ wbf, const float* __restrict__ w) {
    int t = blockIdx.x * blockDim.x + threadIdx.x;   // 128*128
    int c = t >> 7, k = t & 127;
    wbf[t] = (__bf16)w[k * E_DIM + c];               // wbf[c*128 + k]
}

// ---------------- GEMM: O[N,128] = H[N,128] @ W  via bf16 WMMA ------------
// One wave per 16x16 output tile; K=128 in 4 chunks of 32.
// A k order used here (contiguous, as B): lanes 0-15 -> M=0..15, K-low half; lanes 16-31 same M, K-high half.
// B (bf16, col-major in LDS-free global): lane n = lane&15, 16 contiguous K values.

__global__ void __launch_bounds__(256)
k_gemm(const float* __restrict__ H, const __bf16* __restrict__ WBF,
       float* __restrict__ O) {
    const int wave = threadIdx.x >> 5;      // 0..7 -> n-tile
    const int lane = threadIdx.x & 31;
    const int m0   = blockIdx.x * 16;
    const int n0   = wave * 16;
    const int mr   = lane & 15;             // M row (A) / N col (B,C)
    const int half = lane >> 4;

    v8f acc = {};
#pragma unroll
    for (int k0 = 0; k0 < E_DIM; k0 += 32) {
        // ---- A: two contiguous 8-float runs, convert to bf16
        const float* arow = H + (size_t)(m0 + mr) * E_DIM + k0 + half * 16;
        const v8f a0 = *(const v8f*)(arow);        // K = k0 + half*16 + 0..7
        const v8f a1 = *(const v8f*)(arow + 8);    // K = k0 + half*16 + 8..15
        v16bf a;
#pragma unroll
        for (int j = 0; j < 8; ++j) {
            a[j]     = (__bf16)a0[j];
            a[j + 8] = (__bf16)a1[j];
        }
        // ---- B: 16 contiguous bf16 (col-major W), K = k0 + half*16 + 0..15
        const v16bf b = *(const v16bf*)(WBF + (size_t)(n0 + mr) * E_DIM + k0 + half * 16);

        acc = __builtin_amdgcn_wmma_f32_16x16x32_bf16(
            false, a, false, b, (short)0, acc, false, false);
    }
    // ---- C/D: lane -> col n0+mr ; VGPR r -> row m0 + half*8 + r
#pragma unroll
    for (int r = 0; r < 8; ++r)
        O[(size_t)(m0 + half * 8 + r) * E_DIM + n0 + mr] = acc[r];
}

// ---------------- self-loop + bias init: h' = hw * dinv^2 + b -------------

__global__ void k_selfbias(float* __restrict__ hn, const float* __restrict__ hw,
                           const float* __restrict__ dinv, const float* __restrict__ bias) {
    int t = blockIdx.x * blockDim.x + threadIdx.x;   // over N*128/4
    int n = t >> 5, q = t & 31;
    float s = dinv[n]; s = s * s;
    float4 v = *(const float4*)(hw + (size_t)n * E_DIM + q * 4);
    float4 b = *(const float4*)(bias + q * 4);
    float4 o = { v.x * s + b.x, v.y * s + b.y, v.z * s + b.z, v.w * s + b.w };
    *(float4*)(hn + (size_t)n * E_DIM + q * 4) = o;
}

// ---------------- edge scatter: h'[dst] += hw[src] * norm -----------------
// One wave per edge: 512B coalesced gather, scalarized metadata, HW f32 atomics.

__global__ void __launch_bounds__(256)
k_scatter(const float* __restrict__ HW, float* __restrict__ OUT,
          const int* __restrict__ src, const int* __restrict__ dst,
          const float* __restrict__ norm, int ne) {
    const int lane = threadIdx.x & 31;
    int e = __builtin_amdgcn_readfirstlane(blockIdx.x * 8 + (threadIdx.x >> 5));
    if (e >= ne) return;
    const int   s = __builtin_amdgcn_readfirstlane(src[e]);
    const int   d = __builtin_amdgcn_readfirstlane(dst[e]);
    const float w = norm[e];
    const float4 v = *(const float4*)(HW + (size_t)s * E_DIM + lane * 4);
    float* o = OUT + (size_t)d * E_DIM + lane * 4;
    unsafeAtomicAdd(o + 0, v.x * w);
    unsafeAtomicAdd(o + 1, v.y * w);
    unsafeAtomicAdd(o + 2, v.z * w);
    unsafeAtomicAdd(o + 3, v.w * w);
}

// ---------------- global add pool (LDS-staged) ----------------------------

__global__ void k_zero(float* __restrict__ p, int n) {
    int i = blockIdx.x * blockDim.x + threadIdx.x;
    if (i < n) p[i] = 0.0f;
}

__global__ void __launch_bounds__(256)
k_pool(const float* __restrict__ H, const int* __restrict__ batch,
       float* __restrict__ pooled, int nn) {
    __shared__ float acc[N_GRAPHS * E_DIM];          // 16 KB
    for (int i = threadIdx.x; i < N_GRAPHS * E_DIM; i += blockDim.x) acc[i] = 0.0f;
    __syncthreads();
    const int lane = threadIdx.x & 31;
    const int wave = threadIdx.x >> 5;
    const int stride = gridDim.x * 8;
    for (int n = blockIdx.x * 8 + wave; n < nn; n += stride) {
        int g = __builtin_amdgcn_readfirstlane(batch[n]);
        float4 v = *(const float4*)(H + (size_t)n * E_DIM + lane * 4);
        float* a = acc + g * E_DIM + lane * 4;
        atomicAdd(a + 0, v.x);                       // ds_add_f32
        atomicAdd(a + 1, v.y);
        atomicAdd(a + 2, v.z);
        atomicAdd(a + 3, v.w);
    }
    __syncthreads();
    for (int i = threadIdx.x; i < N_GRAPHS * E_DIM; i += blockDim.x) {
        float v = acc[i];
        if (v != 0.0f) unsafeAtomicAdd(&pooled[i], v);
    }
}

// ---------------- readout MLP ---------------------------------------------

__global__ void k_mlp1(float* __restrict__ hr, const float* __restrict__ pooled,
                       const float* __restrict__ w, const float* __restrict__ b) {
    int g = blockIdx.x, j = threadIdx.x;             // 32 blocks x 128 threads
    float a = b[j];
    const float* p = pooled + g * E_DIM;
#pragma unroll 4
    for (int k = 0; k < E_DIM; ++k) a += p[k] * w[k * E_DIM + j];
    hr[g * E_DIM + j] = tanhf(a);
}

__global__ void k_mlp2(float* __restrict__ out, const float* __restrict__ hr,
                       const float* __restrict__ w, const float* __restrict__ b) {
    int g = threadIdx.x;                             // 32 threads
    if (g >= N_GRAPHS) return;
    float a = b[0];
    const float* h = hr + g * E_DIM;
#pragma unroll 4
    for (int j = 0; j < E_DIM; ++j) a += h[j] * w[j];
    out[g] = a;
}

// ---------------------------------------------------------------------------

extern "C" void kernel_launch(void* const* d_in, const int* in_sizes, int n_in,
                              void* d_out, int out_size, void* d_ws, size_t ws_size,
                              hipStream_t stream) {
    const float* x       = (const float*)d_in[0];
    const int*   eidx    = (const int*)  d_in[1];
    const int*   batch   = (const int*)  d_in[2];
    const float* lin_w   = (const float*)d_in[3];
    const float* lin_b   = (const float*)d_in[4];
    const float* gcn_w   = (const float*)d_in[5];
    const float* gcn_b   = (const float*)d_in[6];
    const float* r1_w    = (const float*)d_in[7];
    const float* r1_b    = (const float*)d_in[8];
    const float* r2_w    = (const float*)d_in[9];
    const float* r2_b    = (const float*)d_in[10];
    float* out = (float*)d_out;

    const int* srcI = eidx;             // edge_index[0]
    const int* dstI = eidx + N_EDGES;   // edge_index[1]

    char* ws = (char*)d_ws;
    size_t off = 0;
    auto alloc = [&](size_t bytes) -> void* {
        void* p = ws + off;
        off += (bytes + 255) & ~(size_t)255;
        return p;
    };
    float*  dinv   = (float*) alloc((size_t)N_NODES * 4);
    float*  normw  = (float*) alloc((size_t)N_EDGES * 4);
    float*  hA     = (float*) alloc((size_t)N_NODES * E_DIM * 4);
    float*  hB     = (float*) alloc((size_t)N_NODES * E_DIM * 4);
    __bf16* wbf    = (__bf16*)alloc((size_t)E_DIM * E_DIM * 2);
    float*  pooled = (float*) alloc((size_t)N_GRAPHS * E_DIM * 4);
    float*  hr     = (float*) alloc((size_t)N_GRAPHS * E_DIM * 4);

    const int T = 256;
    // degree / norm (shared by all layers)
    k_fill1 <<<(N_NODES + T - 1) / T, T, 0, stream>>>(dinv, N_NODES);
    k_deg   <<<(N_EDGES + T - 1) / T, T, 0, stream>>>(dinv, dstI, N_EDGES);
    k_rsqrt <<<(N_NODES + T - 1) / T, T, 0, stream>>>(dinv, N_NODES);
    k_norm  <<<(N_EDGES + T - 1) / T, T, 0, stream>>>(normw, srcI, dstI, dinv, N_EDGES);

    // embedding
    k_embed <<<(N_NODES * E_DIM) / T, T, 0, stream>>>(hA, x, lin_w, lin_b);

    // 3 GCN layers
    for (int l = 0; l < 3; ++l) {
        k_wcvt     <<<(E_DIM * E_DIM) / T, T, 0, stream>>>(wbf, gcn_w + (size_t)l * E_DIM * E_DIM);
        k_gemm     <<<N_NODES / 16, T, 0, stream>>>(hA, wbf, hB);
        k_selfbias <<<(N_NODES * E_DIM / 4) / T, T, 0, stream>>>(hA, hB, dinv, gcn_b + (size_t)l * E_DIM);
        k_scatter  <<<(N_EDGES + 7) / 8, T, 0, stream>>>(hB, hA, srcI, dstI, normw, N_EDGES);
    }

    // pool + readout
    k_zero <<<(N_GRAPHS * E_DIM) / T, T, 0, stream>>>(pooled, N_GRAPHS * E_DIM);
    k_pool <<<256, T, 0, stream>>>(hA, batch, pooled, N_NODES);
    k_mlp1 <<<N_GRAPHS, E_DIM, 0, stream>>>(hr, pooled, r1_w, r1_b);
    k_mlp2 <<<1, 32, 0, stream>>>(out, hr, r2_w, r2_b);
}
